// ResidualQuantizer_7052336300674
// MI455X (gfx1250) — compile-verified, matrix operand order checked by reading, not run on hardware
//
#include <hip/hip_runtime.h>

// ---------------------------------------------------------------------------
// Residual VQ for MI455X (gfx1250, wave32, WMMA 16x16x32 bf16)
//   NUM_Q=8 stages, K=1024 codewords, D=512 dim, N=65536 vectors
// ---------------------------------------------------------------------------

typedef __bf16 bf16_t;
typedef bf16_t v16bf __attribute__((ext_vector_type(16)));
typedef bf16_t v8bf  __attribute__((ext_vector_type(8)));
typedef bf16_t v4bf  __attribute__((ext_vector_type(4)));
typedef float  v8f   __attribute__((ext_vector_type(8)));

#define NQ     8
#define KCB    1024
#define DIM    512
#define NROW   65536
#define NTOT   (NROW * DIM)          // 33,554,432

#define ROWS_PER_BLOCK 64            // 4 waves x 16 rows
#define LDS_STRIDE     520           // bf16 elems per row, padded (1040 B -> bank-spread)
#define SMEM_BYTES     (ROWS_PER_BLOCK * LDS_STRIDE * 2 * 2)   // hi + lo arrays, 133,120 B

union V16 { v16bf v; v8bf h[2]; };

struct HL4 { v4bf h; v4bf l; };

__device__ __forceinline__ HL4 split4(float4 v) {
    const bf16_t h0 = (bf16_t)v.x;
    const bf16_t h1 = (bf16_t)v.y;
    const bf16_t h2 = (bf16_t)v.z;
    const bf16_t h3 = (bf16_t)v.w;
    HL4 r;
    r.h.x = h0; r.h.y = h1; r.h.z = h2; r.h.w = h3;
    r.l.x = (bf16_t)(v.x - (float)h0);
    r.l.y = (bf16_t)(v.y - (float)h1);
    r.l.z = (bf16_t)(v.z - (float)h2);
    r.l.w = (bf16_t)(v.w - (float)h3);
    return r;
}

__device__ __forceinline__ v8f wmma_bf16(v16bf a, v16bf b, v8f c) {
    return __builtin_amdgcn_wmma_f32_16x16x32_bf16(false, a, false, b,
                                                   (short)0, c, false, false);
}

// ---------------------------------------------------------------------------
// Prep: split codebooks into interleaved bf16 hi|lo rows, 0.5*||W_k||^2,
// zero rowss.  Layout: cb_split[row][0..511]=hi, [512..1023]=lo.
// grid 1024 x 256: one wave per codeword row (8192 rows).
// ---------------------------------------------------------------------------
__global__ void __launch_bounds__(256) k_prep(const float* __restrict__ cb,
                                              bf16_t* __restrict__ cb_split,
                                              float* __restrict__ halfwn,
                                              double* __restrict__ rowss) {
    const int tid  = blockIdx.x * 256 + threadIdx.x;
    if (tid < NROW) rowss[tid] = 0.0;

    const int wave = threadIdx.x >> 5;
    const int lane = threadIdx.x & 31;
    const int row  = blockIdx.x * 8 + wave;      // 0..8191 (stage*1024 + k)
    const float* src = cb + (size_t)row * DIM;
    bf16_t* dst = cb_split + (size_t)row * (2 * DIM);

    float ss = 0.f;
#pragma unroll
    for (int it = 0; it < 4; ++it) {
        const int c = (lane + it * 32) * 4;
        const float4 v = *(const float4*)(src + c);
        const HL4 s = split4(v);
        *(v4bf*)(dst + c)       = s.h;
        *(v4bf*)(dst + DIM + c) = s.l;
        ss += v.x * v.x + v.y * v.y + v.z * v.z + v.w * v.w;
    }
#pragma unroll
    for (int o = 16; o > 0; o >>= 1) ss += __shfl_xor(ss, o, 32);
    if (lane == 0) halfwn[row] = 0.5f * ss;
}

// ---------------------------------------------------------------------------
// Scores + argmin: idx[n] = argmax_k ( r_n . W_k - 0.5*||W_k||^2 )
// Split-bf16 GEMM: dot ~= hi.hi + lo.hi + hi.lo  (fp32 WMMA accumulate).
// Block: 128 threads (4 waves), 64 rows staged to LDS as hi/lo bf16.
// Each wave: 16 rows x all 1024 codewords, col tiles in groups of 4,
// B registers software-pipelined one k-chunk ahead; A re-read from LDS
// per column-tile group (opaque offset defeats LICM hoisting of A into
// 256 VGPRs, keeping allocation under v256).
// ---------------------------------------------------------------------------
__global__ void __launch_bounds__(128) k_scores(const float* __restrict__ rsrc,
                                                const bf16_t* __restrict__ cb_split,
                                                const float* __restrict__ halfwn,
                                                int* __restrict__ idxbuf,
                                                int stage) {
    extern __shared__ bf16_t smem[];
    bf16_t* sHi = smem;
    bf16_t* sLo = smem + ROWS_PER_BLOCK * LDS_STRIDE;

    const int tid      = threadIdx.x;
    const int lane     = tid & 31;
    const int wave     = tid >> 5;
    const int l16      = lane & 15;
    const int hlf      = lane >> 4;          // 0: K 0..7/16..23, 1: K 8..15/24..31
    const int blockRow = blockIdx.x * ROWS_PER_BLOCK;

    // ---- stage residual rows into LDS with hi/lo split (coalesced) ----
    for (int it = tid; it < ROWS_PER_BLOCK * (DIM / 4); it += 128) {
        const int row = it >> 7;             // / (DIM/4)
        const int c4  = (it & 127) * 4;
        const float4 v = *(const float4*)(rsrc + (size_t)(blockRow + row) * DIM + c4);
        const HL4 s = split4(v);
        *(v4bf*)(sHi + row * LDS_STRIDE + c4) = s.h;
        *(v4bf*)(sLo + row * LDS_STRIDE + c4) = s.l;
    }
    __syncthreads();

    // per-lane A row (A operand: lane l16 holds matrix row M=l16 of this wave's strip)
    const bf16_t* aHiRow = sHi + (wave * 16 + l16) * LDS_STRIDE;
    const bf16_t* aLoRow = sLo + (wave * 16 + l16) * LDS_STRIDE;

    const bf16_t* cbS = cb_split + (size_t)stage * KCB * (2 * DIM);
    const float*  wn  = halfwn + stage * KCB;

    float best[8];
    int   bidx[8];
#pragma unroll
    for (int j = 0; j < 8; ++j) { best[j] = -3.4e38f; bidx[j] = 0; }

    for (int ct = 0; ct < KCB / 16; ct += 4) {           // 16 groups of 4 col tiles
        // Opaque zero offset: formally loop-variant A pointers -> LICM cannot
        // hoist the 256-VGPR A tile set out of this loop.  Offset (not the
        // pointer) is laundered so addrspace inference still emits ds_load.
        int lic = 0;
        asm volatile("" : "+v"(lic));
        const bf16_t* aHiP = aHiRow + lic;
        const bf16_t* aLoP = aLoRow + lic;

        v8f acc[4] = {};
        const bf16_t* bRow[4];
#pragma unroll
        for (int g = 0; g < 4; ++g)
            bRow[g] = cbS + (size_t)((ct + g) * 16 + l16) * (2 * DIM);

        // preload k-chunk 0
        V16 bHc[4], bLc[4];
        {
            const int off0 = hlf * 8;
#pragma unroll
            for (int g = 0; g < 4; ++g) {
                bHc[g].h[0] = *(const v8bf*)(bRow[g] + off0);
                bHc[g].h[1] = *(const v8bf*)(bRow[g] + off0 + 16);
                bLc[g].h[0] = *(const v8bf*)(bRow[g] + DIM + off0);
                bLc[g].h[1] = *(const v8bf*)(bRow[g] + DIM + off0 + 16);
            }
        }

#pragma unroll
        for (int kc = 0; kc < DIM / 32; ++kc) {          // 16 k-chunks of 32
            // prefetch next k-chunk's B tiles (stay in flight during WMMAs)
            V16 bHn[4], bLn[4];
            if (kc < DIM / 32 - 1) {
                const int offn = (kc + 1) * 32 + hlf * 8;
#pragma unroll
                for (int g = 0; g < 4; ++g) {
                    bHn[g].h[0] = *(const v8bf*)(bRow[g] + offn);
                    bHn[g].h[1] = *(const v8bf*)(bRow[g] + offn + 16);
                    bLn[g].h[0] = *(const v8bf*)(bRow[g] + DIM + offn);
                    bLn[g].h[1] = *(const v8bf*)(bRow[g] + DIM + offn + 16);
                }
            }

            const int off = kc * 32 + hlf * 8;
            V16 aH, aL;
            aH.h[0] = *(const v8bf*)(aHiP + off);
            aH.h[1] = *(const v8bf*)(aHiP + off + 16);
            aL.h[0] = *(const v8bf*)(aLoP + off);
            aL.h[1] = *(const v8bf*)(aLoP + off + 16);

            // three split terms, grouped by term: 4 independent acc chains each
#pragma unroll
            for (int g = 0; g < 4; ++g) acc[g] = wmma_bf16(aH.v, bHc[g].v, acc[g]);
#pragma unroll
            for (int g = 0; g < 4; ++g) acc[g] = wmma_bf16(aL.v, bHc[g].v, acc[g]);
#pragma unroll
            for (int g = 0; g < 4; ++g) acc[g] = wmma_bf16(aH.v, bLc[g].v, acc[g]);

            if (kc < DIM / 32 - 1) {
#pragma unroll
                for (int g = 0; g < 4; ++g) { bHc[g] = bHn[g]; bLc[g] = bLn[g]; }
            }
        }

        // score + running argmax (strict > keeps earliest index within a lane)
#pragma unroll
        for (int g = 0; g < 4; ++g) {
            const int cw = (ct + g) * 16 + l16;
            const float hw = wn[cw];
#pragma unroll
            for (int j = 0; j < 8; ++j) {
                const float s = acc[g][j] - hw;
                if (s > best[j]) { best[j] = s; bidx[j] = cw; }
            }
        }
    }

    // cross-lane argmax within each 16-lane half (masks < 16 stay in-half);
    // tie -> lower index, matching argmin's first-match semantics.
#pragma unroll
    for (int j = 0; j < 8; ++j) {
        float v = best[j];
        int   bi = bidx[j];
#pragma unroll
        for (int o = 8; o > 0; o >>= 1) {
            const float ov = __shfl_xor(v, o, 32);
            const int   oi = __shfl_xor(bi, o, 32);
            if (ov > v || (ov == v && oi < bi)) { v = ov; bi = oi; }
        }
        if (l16 == 0) {
            const int row = blockRow + wave * 16 + j + hlf * 8;  // C layout: VGPR j -> M=j / M=j+8
            idxbuf[row] = bi;
        }
    }
}

// ---------------------------------------------------------------------------
// Update: r_new = r_old - W[idx]; rowss[row] += sum(r_new^2) (single writer).
// grid 8192 x 256: one wave per row.
// ---------------------------------------------------------------------------
__global__ void __launch_bounds__(256) k_update(const float* rold,
                                                const float* __restrict__ cb,
                                                const int* __restrict__ idxbuf,
                                                float* resid,
                                                double* __restrict__ rowss,
                                                int stage) {
    const int wave = threadIdx.x >> 5;
    const int lane = threadIdx.x & 31;
    const int row  = blockIdx.x * 8 + wave;
    const int idx  = idxbuf[row];
    const float* qrow = cb + ((size_t)stage * KCB + idx) * DIM;
    const float* rrow = rold + (size_t)row * DIM;
    float* wrow = resid + (size_t)row * DIM;

    float ss = 0.f;
#pragma unroll
    for (int it = 0; it < 4; ++it) {
        const int c = (lane + it * 32) * 4;
        const float4 r = *(const float4*)(rrow + c);
        const float4 q = *(const float4*)(qrow + c);
        float4 nr;
        nr.x = r.x - q.x; nr.y = r.y - q.y; nr.z = r.z - q.z; nr.w = r.w - q.w;
        *(float4*)(wrow + c) = nr;
        ss += nr.x * nr.x + nr.y * nr.y + nr.z * nr.z + nr.w * nr.w;
    }
#pragma unroll
    for (int o = 16; o > 0; o >>= 1) ss += __shfl_xor(ss, o, 32);
    if (lane == 0) rowss[row] += (double)ss;   // exactly one writer per row -> deterministic
}

// ---------------------------------------------------------------------------
// q_sum = x - resid_final
// ---------------------------------------------------------------------------
__global__ void __launch_bounds__(256) k_final(const float* __restrict__ x,
                                               const float* __restrict__ resid,
                                               float* __restrict__ out) {
    const size_t i = ((size_t)blockIdx.x * 256 + threadIdx.x) * 4;
    const float4 a = *(const float4*)(x + i);
    const float4 r = *(const float4*)(resid + i);
    float4 o;
    o.x = a.x - r.x; o.y = a.y - r.y; o.z = a.z - r.z; o.w = a.w - r.w;
    *(float4*)(out + i) = o;
}

// ---------------------------------------------------------------------------
// loss = 1.25 * sum(rowss) / (N*D), deterministic fixed-order reduction.
// ---------------------------------------------------------------------------
__global__ void __launch_bounds__(256) k_loss(const double* __restrict__ rowss,
                                              float* __restrict__ out) {
    __shared__ double sp[256];
    double s = 0.0;
    for (int i = threadIdx.x; i < NROW; i += 256) s += rowss[i];
    sp[threadIdx.x] = s;
    __syncthreads();
    for (int o = 128; o > 0; o >>= 1) {
        if (threadIdx.x < o) sp[threadIdx.x] += sp[threadIdx.x + o];
        __syncthreads();
    }
    if (threadIdx.x == 0) out[NTOT] = 1.25f * (float)(sp[0] / (double)NTOT);
}

// ---------------------------------------------------------------------------
extern "C" void kernel_launch(void* const* d_in, const int* in_sizes, int n_in,
                              void* d_out, int out_size, void* d_ws, size_t ws_size,
                              hipStream_t stream) {
    (void)in_sizes; (void)n_in; (void)out_size; (void)ws_size;
    const float* x  = (const float*)d_in[0];            // [N, D] fp32
    const float* cb = (const float*)d_in[1];            // [8, K, D] fp32
    float* out = (float*)d_out;                         // [N*D] q_sum + [1] loss

    // workspace layout (~152 MB)
    char* ws = (char*)d_ws;
    float*   resid    = (float*)ws;                                   // 134,217,728 B
    bf16_t*  cb_split = (bf16_t*)(ws + (size_t)NROW * DIM * 4);       //  16,777,216 B (hi|lo interleaved)
    float*   halfwn   = (float*)(cb_split + (size_t)NQ * KCB * 2 * DIM); //   32,768 B
    int*     idxbuf   = (int*)(halfwn + NQ * KCB);                    //     262,144 B
    double*  rowss    = (double*)(idxbuf + NROW);                     //     524,288 B

    k_prep<<<dim3(1024), dim3(256), 0, stream>>>(cb, cb_split, halfwn, rowss);

    for (int s = 0; s < NQ; ++s) {
        const float* rsrc = (s == 0) ? x : resid;
        k_scores<<<dim3(NROW / ROWS_PER_BLOCK), dim3(128), SMEM_BYTES, stream>>>(
            rsrc, cb_split, halfwn, idxbuf, s);
        k_update<<<dim3(NROW / 8), dim3(256), 0, stream>>>(
            rsrc, cb, idxbuf, resid, rowss, s);
    }

    k_final<<<dim3(NTOT / 1024), dim3(256), 0, stream>>>(x, resid, out);
    k_loss<<<dim3(1), dim3(256), 0, stream>>>(rowss, out);
}
